// SwinConvNeXtBlock_66889820668064
// MI455X (gfx1250) — matrix operand order repaired and checked by reading, not compiled, not run on hardware
//
#include <hip/hip_runtime.h>
#include <hip/hip_bf16.h>

typedef __attribute__((ext_vector_type(16))) _Float16 v16h;
typedef __attribute__((ext_vector_type(8)))  float    v8f;

#define DIM 192
#define HEADS 6
#define HDIM 32
#define WIN 8
#define SHIFT 4
#define HH 128
#define WW 128
#define BB 8
#define T_TOK 64
#define NW 256
#define TOKENS (BB * HH * WW)          // 131072
#define SCALE 0.17677669529663687f     // 1/sqrt(32)

union Frag { v16h h; uint4 q[2]; };

__device__ __forceinline__ float gelu_exact(float v) {
    return 0.5f * v * (1.0f + erff(v * 0.70710678118654752f));
}

__device__ __forceinline__ int reg3(int g) {   // Swin shift-mask region
    return (g < (HH - WIN)) ? 0 : ((g < (HH - SHIFT)) ? 1 : 2);
}

// ---------------------------------------------------------------------------
// LN over channels + roll(-4,-4) + window partition.  One wave per token.
// out: f16 [B*NW*T, 192] windowed order.
// ---------------------------------------------------------------------------
__global__ __launch_bounds__(256) void ln1_win_kernel(
    const float* __restrict__ x, const float* __restrict__ g,
    const float* __restrict__ b, _Float16* __restrict__ out)
{
    const int wtok = blockIdx.x * 8 + (threadIdx.x >> 5);
    const int lane = threadIdx.x & 31;
    const int bi  = wtok >> 14;
    const int rem = wtok & 16383;
    const int nw  = rem >> 6;
    const int t   = rem & 63;
    const int gh  = (((nw >> 4) << 3) + (t >> 3) + SHIFT) & (HH - 1);
    const int gw  = (((nw & 15) << 3) + (t & 7) + SHIFT) & (WW - 1);
    const float* px = x + (((size_t)bi * DIM) << 14) + (gh << 7) + gw;
    float v[6], s = 0.f, sq = 0.f;
#pragma unroll
    for (int j = 0; j < 6; ++j) {
        int c = lane + 32 * j;
        v[j] = px[(size_t)c << 14];
        s += v[j]; sq += v[j] * v[j];
    }
#pragma unroll
    for (int d = 1; d < 32; d <<= 1) { s += __shfl_xor(s, d); sq += __shfl_xor(sq, d); }
    float mean = s * (1.f / DIM);
    float var  = sq * (1.f / DIM) - mean * mean;
    float inv  = rsqrtf(var + 1e-6f);
#pragma unroll
    for (int j = 0; j < 6; ++j) {
        int c = lane + 32 * j;
        out[(size_t)wtok * DIM + c] = (_Float16)((v[j] - mean) * inv * g[c] + b[c]);
    }
}

// LN over channels, plain (b,h,w) token order.
__global__ __launch_bounds__(256) void ln_plain_kernel(
    const float* __restrict__ x, const float* __restrict__ g,
    const float* __restrict__ b, _Float16* __restrict__ out)
{
    const int tok  = blockIdx.x * 8 + (threadIdx.x >> 5);
    const int lane = threadIdx.x & 31;
    const int bi = tok >> 14;
    const int hw = tok & 16383;
    const float* px = x + (((size_t)bi * DIM) << 14) + hw;
    float v[6], s = 0.f, sq = 0.f;
#pragma unroll
    for (int j = 0; j < 6; ++j) {
        int c = lane + 32 * j;
        v[j] = px[(size_t)c << 14];
        s += v[j]; sq += v[j] * v[j];
    }
#pragma unroll
    for (int d = 1; d < 32; d <<= 1) { s += __shfl_xor(s, d); sq += __shfl_xor(sq, d); }
    float mean = s * (1.f / DIM);
    float var  = sq * (1.f / DIM) - mean * mean;
    float inv  = rsqrtf(var + 1e-6f);
#pragma unroll
    for (int j = 0; j < 6; ++j) {
        int c = lane + 32 * j;
        out[(size_t)tok * DIM + c] = (_Float16)((v[j] - mean) * inv * g[c] + b[c]);
    }
}

// ---------------------------------------------------------------------------
// WMMA GEMM: C[M,N] = A[M,K](f16) * Bw[K,N](fp32 -> f16) + bias
// Block = 256 thr (8 waves), 128x64 C tile, each wave 4 (2x2) WMMA tiles.
// RMODE: 0 = write f16 row-major [M,N]; 1 = windowed->NCHW + residual (fp32);
//        2 = plain->NCHW + residual (fp32).
// ---------------------------------------------------------------------------
template <bool GELU_ON, int RMODE, bool GAMMA_ON>
__global__ __launch_bounds__(256) void gemm_wmma_kernel(
    const _Float16* __restrict__ A, const float* __restrict__ Bw,
    const float* __restrict__ bias, void* __restrict__ outp,
    const float* __restrict__ resid, const float* __restrict__ gamma,
    int M, int N, int K)
{
    __shared__ alignas(16) _Float16 As[128][32];
    __shared__ alignas(16) _Float16 Bt[64][32];   // [n][k]
    const int tid  = threadIdx.x;
    const int lane = tid & 31;
    const int wid  = tid >> 5;
    const int lo = lane & 15, hi = lane >> 4;
    const int m0 = blockIdx.x * 128;
    const int n0 = blockIdx.y * 64;
    const int ri = (wid & 3) * 2;        // 2 row tiles of 16
    const int cj = (wid >> 2) * 2;       // 2 col tiles of 16

    v8f acc[2][2] = {{v8f{}, v8f{}}, {v8f{}, v8f{}}};

    for (int k0 = 0; k0 < K; k0 += 32) {
        {   // A tile: 128x32 halves, 16 per thread (2x b128)
            int row = tid >> 1;
            int kc  = (tid & 1) * 16;
            const _Float16* src = &A[(size_t)(m0 + row) * K + k0 + kc];
            *(uint4*)&As[row][kc]     = *(const uint4*)(src);
            *(uint4*)&As[row][kc + 8] = *(const uint4*)(src + 8);
            if (k0 + 32 < K) __builtin_prefetch(src + 32, 0, 1);
        }
        {   // B tile: 32(k) x 64(n) fp32 -> f16, stored n-major
            int kk = tid >> 3;
            int nb = (tid & 7) * 8;
            const float* src = &Bw[(size_t)(k0 + kk) * N + n0 + nb];
#pragma unroll
            for (int j = 0; j < 8; ++j) Bt[nb + j][kk] = (_Float16)src[j];
        }
        __syncthreads();
        Frag fa0, fa1, fb0, fb1;
        fa0.q[0] = *(const uint4*)&As[ri * 16 + lo][hi * 8];
        fa0.q[1] = *(const uint4*)&As[ri * 16 + lo][16 + hi * 8];
        fa1.q[0] = *(const uint4*)&As[(ri + 1) * 16 + lo][hi * 8];
        fa1.q[1] = *(const uint4*)&As[(ri + 1) * 16 + lo][16 + hi * 8];
        fb0.q[0] = *(const uint4*)&Bt[cj * 16 + lo][hi * 8];
        fb0.q[1] = *(const uint4*)&Bt[cj * 16 + lo][16 + hi * 8];
        fb1.q[0] = *(const uint4*)&Bt[(cj + 1) * 16 + lo][hi * 8];
        fb1.q[1] = *(const uint4*)&Bt[(cj + 1) * 16 + lo][16 + hi * 8];
        acc[0][0] = __builtin_amdgcn_wmma_f32_16x16x32_f16(false, fa0.h, false, fb0.h,
                                                           (short)0, acc[0][0], false, false);
        acc[0][1] = __builtin_amdgcn_wmma_f32_16x16x32_f16(false, fa0.h, false, fb1.h,
                                                           (short)0, acc[0][1], false, false);
        acc[1][0] = __builtin_amdgcn_wmma_f32_16x16x32_f16(false, fa1.h, false, fb0.h,
                                                           (short)0, acc[1][0], false, false);
        acc[1][1] = __builtin_amdgcn_wmma_f32_16x16x32_f16(false, fa1.h, false, fb1.h,
                                                           (short)0, acc[1][1], false, false);
        __syncthreads();
    }

#pragma unroll
    for (int u = 0; u < 2; ++u) {
#pragma unroll
        for (int vv = 0; vv < 2; ++vv) {
            const int n = n0 + (cj + vv) * 16 + lo;
            const float bvl = bias[n];
            const float gvl = GAMMA_ON ? gamma[n] : 1.f;
#pragma unroll
            for (int r = 0; r < 8; ++r) {
                const int m = m0 + (ri + u) * 16 + r + hi * 8;
                float v = acc[u][vv][r] + bvl;
                if (GELU_ON) v = gelu_exact(v);
                if (GAMMA_ON) v *= gvl;
                if (RMODE == 0) {
                    ((_Float16*)outp)[(size_t)m * N + n] = (_Float16)v;
                } else if (RMODE == 1) {
                    int bi = m >> 14, rem = m & 16383, nw = rem >> 6, t = rem & 63;
                    int gh = (((nw >> 4) << 3) + (t >> 3) + SHIFT) & (HH - 1);
                    int gw = (((nw & 15) << 3) + (t & 7) + SHIFT) & (WW - 1);
                    size_t idx = (((size_t)bi * DIM + n) << 14) + (gh << 7) + gw;
                    ((float*)outp)[idx] = resid[idx] + v;
                } else {
                    int bi = m >> 14, hw = m & 16383;
                    size_t idx = (((size_t)bi * DIM + n) << 14) + hw;
                    ((float*)outp)[idx] = resid[idx] + v;
                }
            }
        }
    }
}

// ---------------------------------------------------------------------------
// Fused shifted-window attention: one block per (window, head), 4 waves.
// qkv: f16 [BW*64, 576] (q|k|v per head).  out: f16 [BW*64, 192].
// ---------------------------------------------------------------------------
__global__ __launch_bounds__(128) void attn_kernel(
    const _Float16* __restrict__ qkv, _Float16* __restrict__ outp)
{
    __shared__ alignas(16) _Float16 Qs[64][32];
    __shared__ alignas(16) _Float16 Ks[64][32];
    __shared__ alignas(16) _Float16 Vt[32][64];      // [d][s]
    __shared__ alignas(16) _Float16 Ps[4][16][64];   // per-wave prob strip
    const int bw   = blockIdx.x;      // 0..2047
    const int head = blockIdx.y;      // 0..5
    const int tid  = threadIdx.x;
    const int lane = tid & 31, wid = tid >> 5;
    const int lo = lane & 15, hi = lane >> 4;
    const size_t base = (size_t)bw * T_TOK * (3 * DIM);

    {   // stage Q, K, V for this (window, head)
        int s = tid >> 1;
        int c = (tid & 1) * 16;
        const _Float16* qrow = qkv + base + (size_t)s * (3 * DIM) + head * HDIM;
        const _Float16* krow = qrow + DIM;
        const _Float16* vrow = qrow + 2 * DIM;
        *(uint4*)&Qs[s][c]     = *(const uint4*)(qrow + c);
        *(uint4*)&Qs[s][c + 8] = *(const uint4*)(qrow + c + 8);
        *(uint4*)&Ks[s][c]     = *(const uint4*)(krow + c);
        *(uint4*)&Ks[s][c + 8] = *(const uint4*)(krow + c + 8);
#pragma unroll
        for (int j = 0; j < 16; ++j) Vt[c + j][s] = vrow[c + j];
    }
    __syncthreads();

    const int tbase = wid * 16;
    Frag fq;
    fq.q[0] = *(const uint4*)&Qs[tbase + lo][hi * 8];
    fq.q[1] = *(const uint4*)&Qs[tbase + lo][16 + hi * 8];

    v8f sc[4] = {v8f{}, v8f{}, v8f{}, v8f{}};
#pragma unroll
    for (int kb = 0; kb < 4; ++kb) {
        Frag fk;
        fk.q[0] = *(const uint4*)&Ks[kb * 16 + lo][hi * 8];
        fk.q[1] = *(const uint4*)&Ks[kb * 16 + lo][16 + hi * 8];
        sc[kb] = __builtin_amdgcn_wmma_f32_16x16x32_f16(false, fq.h, false, fk.h,
                                                        (short)0, sc[kb], false, false);
    }

    // scale + shift-window mask + softmax over 64 keys per row
    const int nw = bw & (NW - 1);
    const int wy = nw >> 4, wx = nw & 15;
    int rids[4], ridt[8];
#pragma unroll
    for (int kb = 0; kb < 4; ++kb) {
        int s = kb * 16 + lo;
        rids[kb] = reg3(wy * 8 + (s >> 3)) * 3 + reg3(wx * 8 + (s & 7));
    }
#pragma unroll
    for (int r = 0; r < 8; ++r) {
        int t = tbase + r + hi * 8;
        ridt[r] = reg3(wy * 8 + (t >> 3)) * 3 + reg3(wx * 8 + (t & 7));
    }
    float p[4][8];
#pragma unroll
    for (int kb = 0; kb < 4; ++kb)
#pragma unroll
        for (int r = 0; r < 8; ++r)
            p[kb][r] = sc[kb][r] * SCALE + ((rids[kb] == ridt[r]) ? 0.f : -100.f);

#pragma unroll
    for (int r = 0; r < 8; ++r) {
        float mx = fmaxf(fmaxf(p[0][r], p[1][r]), fmaxf(p[2][r], p[3][r]));
#pragma unroll
        for (int d = 1; d < 16; d <<= 1) mx = fmaxf(mx, __shfl_xor(mx, d));
        float sum = 0.f;
#pragma unroll
        for (int kb = 0; kb < 4; ++kb) { p[kb][r] = __expf(p[kb][r] - mx); sum += p[kb][r]; }
#pragma unroll
        for (int d = 1; d < 16; d <<= 1) sum += __shfl_xor(sum, d);
        float inv = 1.f / sum;
        int m = r + hi * 8;
#pragma unroll
        for (int kb = 0; kb < 4; ++kb)
            Ps[wid][m][kb * 16 + lo] = (_Float16)(p[kb][r] * inv);
    }
    // (same-wave LDS RAW — DScnt ordering inserted by compiler)

    v8f o0 = {}, o1 = {};
#pragma unroll
    for (int kc = 0; kc < 2; ++kc) {
        Frag fp, fv0, fv1;
        fp.q[0]  = *(const uint4*)&Ps[wid][lo][kc * 32 + hi * 8];
        fp.q[1]  = *(const uint4*)&Ps[wid][lo][kc * 32 + 16 + hi * 8];
        fv0.q[0] = *(const uint4*)&Vt[lo][kc * 32 + hi * 8];
        fv0.q[1] = *(const uint4*)&Vt[lo][kc * 32 + 16 + hi * 8];
        fv1.q[0] = *(const uint4*)&Vt[16 + lo][kc * 32 + hi * 8];
        fv1.q[1] = *(const uint4*)&Vt[16 + lo][kc * 32 + 16 + hi * 8];
        o0 = __builtin_amdgcn_wmma_f32_16x16x32_f16(false, fp.h, false, fv0.h,
                                                    (short)0, o0, false, false);
        o1 = __builtin_amdgcn_wmma_f32_16x16x32_f16(false, fp.h, false, fv1.h,
                                                    (short)0, o1, false, false);
    }
#pragma unroll
    for (int r = 0; r < 8; ++r) {
        size_t row = (size_t)(bw * T_TOK + tbase + r + hi * 8) * DIM + head * HDIM;
        outp[row + lo]      = (_Float16)o0[r];
        outp[row + 16 + lo] = (_Float16)o1[r];
    }
}

// ---------------------------------------------------------------------------
// Depthwise 7x7 conv, NCHW, SAME zero padding.
// ---------------------------------------------------------------------------
__global__ __launch_bounds__(256) void dwconv_kernel(
    const float* __restrict__ x, const float* __restrict__ k7,
    const float* __restrict__ kb, float* __restrict__ out)
{
    int idx = blockIdx.x * 256 + threadIdx.x;
    if (idx >= BB * DIM * HH * WW) return;
    int hw = idx & 16383;
    int c  = (idx >> 14) % DIM;
    int bi = idx / (DIM * HH * WW);
    int h = hw >> 7, w = hw & 127;
    const float* plane = x + (((size_t)bi * DIM + c) << 14);
    float s = 0.f;
#pragma unroll
    for (int ky = 0; ky < 7; ++ky) {
        int yy = h + ky - 3;
        if (yy < 0 || yy >= HH) continue;
#pragma unroll
        for (int kx = 0; kx < 7; ++kx) {
            int xx = w + kx - 3;
            if (xx < 0 || xx >= WW) continue;
            s += plane[(yy << 7) + xx] * k7[(ky * 7 + kx) * DIM + c];
        }
    }
    out[idx] = s + kb[c];
}

// ---------------------------------------------------------------------------
extern "C" void kernel_launch(void* const* d_in, const int* in_sizes, int n_in,
                              void* d_out, int out_size, void* d_ws, size_t ws_size,
                              hipStream_t stream) {
    const float* x      = (const float*)d_in[0];
    const float* g1     = (const float*)d_in[1];
    const float* b1     = (const float*)d_in[2];
    const float* qkv_w  = (const float*)d_in[3];
    const float* qkv_b  = (const float*)d_in[4];
    const float* proj_w = (const float*)d_in[5];
    const float* proj_b = (const float*)d_in[6];
    const float* g2     = (const float*)d_in[7];
    const float* b2     = (const float*)d_in[8];
    const float* mlp_w1 = (const float*)d_in[9];
    const float* mlp_b1 = (const float*)d_in[10];
    const float* mlp_w2 = (const float*)d_in[11];
    const float* mlp_b2 = (const float*)d_in[12];
    const float* dw_k   = (const float*)d_in[13];
    const float* dw_b   = (const float*)d_in[14];
    const float* cn_g   = (const float*)d_in[15];
    const float* cn_b   = (const float*)d_in[16];
    const float* pw1_w  = (const float*)d_in[17];
    const float* pw1_b  = (const float*)d_in[18];
    const float* pw2_w  = (const float*)d_in[19];
    const float* pw2_b  = (const float*)d_in[20];
    const float* ls_g   = (const float*)d_in[21];

    char* ws = (char*)d_ws;
    const size_t szA = (size_t)TOKENS * DIM * sizeof(_Float16);      // 50.3 MB
    const size_t szB = (size_t)TOKENS * 4 * DIM * sizeof(_Float16);  // 201 MB
    const size_t szF = (size_t)TOKENS * DIM * sizeof(float);         // 100 MB
    _Float16* bufA = (_Float16*)(ws);                 // LN out / attn out / LN out
    _Float16* bufB = (_Float16*)(ws + szA);           // qkv / mlp hidden / pw hidden
    float*    x1   = (float*)(ws + szA + szB);        // attn residual result (NCHW)
    float*    x2   = (float*)(ws + szA + szB + szF);  // mlp residual result  (NCHW)
    float*    dwo  = x1;                              // x1 dead after mlp2 -> reuse

    const dim3 blk256(256), blk128(128);
    const int lnBlocks = TOKENS / 8;                  // 16384
    const int gx = TOKENS / 128;                      // 1024 row-tiles

    // 1) LN1 + roll + window partition -> bufA (f16, windowed order)
    ln1_win_kernel<<<lnBlocks, blk256, 0, stream>>>(x, g1, b1, bufA);
    // 2) QKV GEMM: [131072,192] x [192,576] -> bufB (f16)
    gemm_wmma_kernel<false, 0, false><<<dim3(gx, 9), blk256, 0, stream>>>(
        bufA, qkv_w, qkv_b, bufB, nullptr, nullptr, TOKENS, 3 * DIM, DIM);
    // 3) fused window attention -> bufA (f16)
    attn_kernel<<<dim3(BB * NW, HEADS), blk128, 0, stream>>>(bufB, bufA);
    // 4) proj GEMM + un-window/un-roll + residual(x) -> x1 (fp32 NCHW)
    gemm_wmma_kernel<false, 1, false><<<dim3(gx, 3), blk256, 0, stream>>>(
        bufA, proj_w, proj_b, x1, x, nullptr, TOKENS, DIM, DIM);
    // 5) LN2 -> bufA (f16, plain order)
    ln_plain_kernel<<<lnBlocks, blk256, 0, stream>>>(x1, g2, b2, bufA);
    // 6) MLP fc1 + GELU -> bufB (f16)
    gemm_wmma_kernel<true, 0, false><<<dim3(gx, 12), blk256, 0, stream>>>(
        bufA, mlp_w1, mlp_b1, bufB, nullptr, nullptr, TOKENS, 4 * DIM, DIM);
    // 7) MLP fc2 + residual(x1) -> x2 (fp32 NCHW)
    gemm_wmma_kernel<false, 2, false><<<dim3(gx, 3), blk256, 0, stream>>>(
        bufB, mlp_w2, mlp_b2, x2, x1, nullptr, TOKENS, DIM, 4 * DIM);
    // 8) depthwise 7x7 conv -> dwo (fp32 NCHW, reuses x1)
    dwconv_kernel<<<(BB * DIM * HH * WW) / 256, blk256, 0, stream>>>(x2, dw_k, dw_b, dwo);
    // 9) ConvNeXt LN -> bufA (f16)
    ln_plain_kernel<<<lnBlocks, blk256, 0, stream>>>(dwo, cn_g, cn_b, bufA);
    // 10) pw1 + GELU -> bufB (f16)
    gemm_wmma_kernel<true, 0, false><<<dim3(gx, 12), blk256, 0, stream>>>(
        bufA, pw1_w, pw1_b, bufB, nullptr, nullptr, TOKENS, 4 * DIM, DIM);
    // 11) pw2 + layer-scale gamma + residual(x2) -> d_out (fp32 NCHW)
    gemm_wmma_kernel<false, 2, true><<<dim3(gx, 3), blk256, 0, stream>>>(
        bufB, pw2_w, pw2_b, (float*)d_out, x2, ls_g, TOKENS, DIM, 4 * DIM);
    (void)in_sizes; (void)n_in; (void)out_size; (void)ws_size;
}
